// InnerAttention_6012954214874
// MI455X (gfx1250) — matrix operand order repaired and checked by reading, not compiled, not run on hardware
//
#include <hip/hip_runtime.h>
#include <math.h>

typedef __attribute__((ext_vector_type(16))) _Float16 v16h;
typedef __attribute__((ext_vector_type(8)))  float    v8f;

// Problem constants (from reference)
constexpr int Bsz  = 32;
constexpr int Nseq = 512;
constexpr int Cdim = 512;
constexpr int H    = 8;
constexpr int Dh   = 64;
constexpr int QKVN = 3 * H * Dh;   // 1536
constexpr float SCALE = 0.125f;    // Dh^-0.5

// ---- wave32 WMMA fragment maps used here (f16 16x16x32; contiguous k, same for A and B) -------------------
// A matrix 16x32 (MxK): lane L holds row M=L&15,
//   half h -> K = ((L>=16)?16:0) + h
__device__ __forceinline__ int kmapA(int lane, int h) {
    return ((lane >> 4) << 4) + h;
}
// B matrix 32x16 (KxN): lane L holds col N=L&15, K = ((L>=16)?16:0) + h
__device__ __forceinline__ int kbaseB(int lane) { return (lane >> 4) << 4; }
// C/D 16x16 f32: lane L, vgpr r -> M = r + ((L>=16)?8:0), N = L&15

#define WMMA_F16(a, b, c) \
    __builtin_amdgcn_wmma_f32_16x16x32_f16(false, (a), false, (b), (short)0, (c), false, false)

// =============================================================================
// Kernel 1: qkv = x @ qkv_w^T + qkv_b ; split into q(scaled)/k/v as f16 [B,H,N,D]
// Each wave computes a 32x64 tile (2x4 accumulators): 8 WMMA per 6 fragment
// loads per K-step.
// =============================================================================
__global__ __launch_bounds__(256) void qkv_gemm_kernel(
    const float* __restrict__ x, const float* __restrict__ w,
    const float* __restrict__ bias,
    _Float16* __restrict__ qh, _Float16* __restrict__ kh, _Float16* __restrict__ vh)
{
    const int lane = threadIdx.x & 31;
    const int wave = threadIdx.x >> 5;
    const int tile = blockIdx.x * 8 + wave;      // 12288 wave-tiles total
    const int nT = QKVN / 64;                    // 24
    const int m0 = (tile / nT) * 32;             // row in [0, B*N)
    const int j0 = (tile % nT) * 64;             // col in [0, 1536)
    const int li = lane & 15, hi8 = (lane >> 4) << 3;

    v8f acc[2][4] = {};
    for (int k0 = 0; k0 < Cdim; k0 += 32) {
        // gfx1250 prefetch of the next K-slab (speculative is fine at the edge)
        __builtin_prefetch(&x[(m0 + li) * Cdim + k0 + 32], 0, 1);
        __builtin_prefetch(&w[(j0 + li) * Cdim + k0 + 32], 0, 1);
        v16h av[2], bv[4];
#pragma unroll
        for (int mi = 0; mi < 2; ++mi)
#pragma unroll
            for (int h = 0; h < 16; ++h)
                av[mi][h] = (_Float16)x[(m0 + mi * 16 + li) * Cdim + k0 + kmapA(lane, h)];
#pragma unroll
        for (int ji = 0; ji < 4; ++ji)
#pragma unroll
            for (int h = 0; h < 16; ++h)
                bv[ji][h] = (_Float16)w[(j0 + ji * 16 + li) * Cdim + k0 + kbaseB(lane) + h];
#pragma unroll
        for (int mi = 0; mi < 2; ++mi)
#pragma unroll
            for (int ji = 0; ji < 4; ++ji)
                acc[mi][ji] = WMMA_F16(av[mi], bv[ji], acc[mi][ji]);
    }

#pragma unroll
    for (int ji = 0; ji < 4; ++ji) {
        const int j = j0 + ji * 16 + li;
        const float bj = bias[j];
        const int p  = j >> 9;          // 0=q,1=k,2=v (uniform per 16-col subtile)
        const int hh = (j >> 6) & 7;    // head
        const int d  = j & 63;
        _Float16* dst = (p == 0) ? qh : (p == 1) ? kh : vh;
        const float s = (p == 0) ? SCALE : 1.0f;
#pragma unroll
        for (int mi = 0; mi < 2; ++mi)
#pragma unroll
            for (int r = 0; r < 8; ++r) {
                const int m = m0 + mi * 16 + r + hi8;
                const int bi = m >> 9, n = m & 511;
                dst[((bi * H + hh) * Nseq + n) * Dh + d] = (_Float16)((acc[mi][ji][r] + bj) * s);
            }
    }
}

// =============================================================================
// Kernel 2: fused attention per (b, h, 64-query tile).
// LDS: Qs 80x64 f16 | Vs 512x64 f16 | Ss 80x512 f32 | Ls 64x512 f16 | red
// S = Q@K^T -> EPEG 15-tap conv over query axis (+bias) -> softmax -> P@V
// =============================================================================
constexpr int RS = 80;  // 64 rows + 8 halo each side
constexpr size_t OFF_VS  = (size_t)RS * Dh * sizeof(_Float16);               // 10240
constexpr size_t OFF_SS  = OFF_VS + (size_t)Nseq * Dh * sizeof(_Float16);    // +65536
constexpr size_t OFF_LS  = OFF_SS + (size_t)RS * 512 * sizeof(float);        // +163840
constexpr size_t OFF_RED = OFF_LS + (size_t)64 * 512 * sizeof(_Float16);     // +65536
constexpr size_t SMEM_BYTES = OFF_RED + 256 * sizeof(float);                 // 306176 B

__global__ __launch_bounds__(256) void attn_fused_kernel(
    const _Float16* __restrict__ qh, const _Float16* __restrict__ kh,
    const _Float16* __restrict__ vh,
    const float* __restrict__ pe_w, const float* __restrict__ pe_b,
    _Float16* __restrict__ oy)
{
    extern __shared__ char smem[];
    _Float16* Qs = (_Float16*)smem;                  // 80 x 64
    _Float16* Vs = (_Float16*)(smem + OFF_VS);       // 512 x 64
    float*    Ss = (float*)(smem + OFF_SS);          // 80 x 512
    _Float16* Ls = (_Float16*)(smem + OFF_LS);       // 64 x 512
    float*    red = (float*)(smem + OFF_RED);        // 256

    const int tid  = threadIdx.x;
    const int lane = tid & 31, wave = tid >> 5;
    const int qt = blockIdx.x & 7;
    const int bh = blockIdx.x >> 3;          // b*H + h
    const int h  = bh & 7;
    const int n0 = qt * 64;
    const int li = lane & 15, hi8 = (lane >> 4) << 3;

    // ---- Phase 1: stage Q tile+halo (zero-padded) and the V panel into LDS ----
    for (int idx = tid; idx < RS * Dh; idx += 256) {
        const int r = idx >> 6, d = idx & 63;
        const int n = n0 - 8 + r;
        Qs[idx] = (n >= 0 && n < Nseq) ? qh[(bh * Nseq + n) * Dh + d] : (_Float16)0.0f;
    }
    {   // V panel: 512x64 f16 = 64KB, vectorized 16B copies
        const float4* src = (const float4*)(vh + (size_t)bh * Nseq * Dh);
        float4* dst = (float4*)Vs;
        for (int idx = tid; idx < Nseq * Dh / 8; idx += 256) dst[idx] = src[idx];
    }
    __syncthreads();

    // ---- Phase 2: S(80x512) = Q @ K^T via WMMA (1x4 column blocking) ----
    for (int u = wave; u < 5 * 8; u += 8) {
        const int i0 = (u >> 3) * 16;      // score row tile
        const int j0 = (u & 7) * 64;       // key col group
        v8f acc[4] = {};
        for (int k0 = 0; k0 < Dh; k0 += 32) {
            v16h a, bv[4];
#pragma unroll
            for (int hh = 0; hh < 16; ++hh)
                a[hh] = Qs[(i0 + li) * Dh + k0 + kmapA(lane, hh)];
#pragma unroll
            for (int ji = 0; ji < 4; ++ji)
#pragma unroll
                for (int hh = 0; hh < 16; ++hh)
                    bv[ji][hh] = kh[(bh * Nseq + j0 + ji * 16 + li) * Dh + k0 + kbaseB(lane) + hh];
#pragma unroll
            for (int ji = 0; ji < 4; ++ji)
                acc[ji] = WMMA_F16(a, bv[ji], acc[ji]);
        }
#pragma unroll
        for (int ji = 0; ji < 4; ++ji)
#pragma unroll
            for (int r = 0; r < 8; ++r)
                Ss[(i0 + r + hi8) * 512 + j0 + ji * 16 + li] = acc[ji][r];
    }
    __syncthreads();

    // ---- Phase 3: logits = S + depthwise conv over query axis + pe_b ----
    float wt[15];
#pragma unroll
    for (int t = 0; t < 15; ++t) wt[t] = pe_w[h * 15 + t];
    const float pb = pe_b[h];
    for (int idx = tid; idx < 64 * 512; idx += 256) {
        const int r = idx >> 9, m = idx & 511;
        const int rs = r + 8;               // center row in Ss
        float v = Ss[rs * 512 + m] + pb;
#pragma unroll
        for (int t = 0; t < 15; ++t) v += wt[t] * Ss[(rs + t - 7) * 512 + m];
        Ls[idx] = (_Float16)v;
    }
    __syncthreads();

    // ---- Phase 4: row softmax over 512 keys (4 threads per row) ----
    {
        const int r = tid >> 2, sub = tid & 3;
        float mx = -1e30f;
        for (int m = sub; m < 512; m += 4) mx = fmaxf(mx, (float)Ls[r * 512 + m]);
        red[tid] = mx;
        __syncthreads();
        mx = fmaxf(fmaxf(red[r * 4 + 0], red[r * 4 + 1]),
                   fmaxf(red[r * 4 + 2], red[r * 4 + 3]));
        __syncthreads();
        float sum = 0.0f;
        for (int m = sub; m < 512; m += 4) sum += __expf((float)Ls[r * 512 + m] - mx);
        red[tid] = sum;
        __syncthreads();
        sum = red[r * 4 + 0] + red[r * 4 + 1] + red[r * 4 + 2] + red[r * 4 + 3];
        const float inv = 1.0f / sum;
        for (int m = sub; m < 512; m += 4)
            Ls[r * 512 + m] = (_Float16)(__expf((float)Ls[r * 512 + m] - mx) * inv);
    }
    __syncthreads();

    // ---- Phase 5: O(64x64) = P @ V via WMMA (A reused across 2 col tiles) ----
    {
        const int rt = wave & 3;                 // output row tile
        const int cb = (wave >> 2) * 2;          // first of 2 col tiles
        const int i0 = rt * 16;
        v8f acc0 = {}, acc1 = {};
        for (int k0 = 0; k0 < 512; k0 += 32) {
            v16h a, b0, b1;
#pragma unroll
            for (int hh = 0; hh < 16; ++hh)
                a[hh] = Ls[(i0 + li) * 512 + k0 + kmapA(lane, hh)];
#pragma unroll
            for (int hh = 0; hh < 16; ++hh) {
                const int kk = k0 + kbaseB(lane) + hh;
                b0[hh] = Vs[kk * Dh + (cb + 0) * 16 + li];
                b1[hh] = Vs[kk * Dh + (cb + 1) * 16 + li];
            }
            acc0 = WMMA_F16(a, b0, acc0);
            acc1 = WMMA_F16(a, b1, acc1);
        }
        const int bidx = bh >> 3;
#pragma unroll
        for (int r = 0; r < 8; ++r) {
            const size_t row = (size_t)bidx * Nseq + n0 + i0 + r + hi8;
            oy[row * (H * Dh) + h * Dh + (cb + 0) * 16 + li] = (_Float16)acc0[r];
            oy[row * (H * Dh) + h * Dh + (cb + 1) * 16 + li] = (_Float16)acc1[r];
        }
    }
}

// =============================================================================
// Kernel 3: out = oy @ proj_w^T + proj_b  (fp32 output)
// Each wave computes a 32x64 tile (2x4 accumulators).
// =============================================================================
__global__ __launch_bounds__(256) void proj_gemm_kernel(
    const _Float16* __restrict__ oy, const float* __restrict__ w,
    const float* __restrict__ bias, float* __restrict__ out)
{
    const int lane = threadIdx.x & 31;
    const int wave = threadIdx.x >> 5;
    const int tile = blockIdx.x * 8 + wave;      // 4096 wave-tiles
    const int nT = Cdim / 64;                    // 8
    const int m0 = (tile / nT) * 32;
    const int j0 = (tile % nT) * 64;
    const int li = lane & 15, hi8 = (lane >> 4) << 3;

    v8f acc[2][4] = {};
    for (int k0 = 0; k0 < H * Dh; k0 += 32) {
        __builtin_prefetch(&oy[(size_t)(m0 + li) * (H * Dh) + k0 + 32], 0, 1);
        __builtin_prefetch(&w[(j0 + li) * (H * Dh) + k0 + 32], 0, 1);
        v16h av[2], bv[4];
#pragma unroll
        for (int mi = 0; mi < 2; ++mi)
#pragma unroll
            for (int h = 0; h < 16; ++h)
                av[mi][h] = oy[(size_t)(m0 + mi * 16 + li) * (H * Dh) + k0 + kmapA(lane, h)];
#pragma unroll
        for (int ji = 0; ji < 4; ++ji)
#pragma unroll
            for (int h = 0; h < 16; ++h)
                bv[ji][h] = (_Float16)w[(j0 + ji * 16 + li) * (H * Dh) + k0 + kbaseB(lane) + h];
#pragma unroll
        for (int mi = 0; mi < 2; ++mi)
#pragma unroll
            for (int ji = 0; ji < 4; ++ji)
                acc[mi][ji] = WMMA_F16(av[mi], bv[ji], acc[mi][ji]);
    }
#pragma unroll
    for (int ji = 0; ji < 4; ++ji) {
        const int j = j0 + ji * 16 + li;
        const float bj = bias[j];
#pragma unroll
        for (int mi = 0; mi < 2; ++mi)
#pragma unroll
            for (int r = 0; r < 8; ++r)
                out[(size_t)(m0 + mi * 16 + r + hi8) * Cdim + j] = acc[mi][ji][r] + bj;
    }
}

// =============================================================================
// Host launcher
// =============================================================================
extern "C" void kernel_launch(void* const* d_in, const int* in_sizes, int n_in,
                              void* d_out, int out_size, void* d_ws, size_t ws_size,
                              hipStream_t stream) {
    const float* x      = (const float*)d_in[0];
    const float* qkv_w  = (const float*)d_in[1];
    const float* qkv_b  = (const float*)d_in[2];
    const float* pe_w   = (const float*)d_in[3];
    const float* pe_b   = (const float*)d_in[4];
    const float* proj_w = (const float*)d_in[5];
    const float* proj_b = (const float*)d_in[6];
    float* out = (float*)d_out;

    // Workspace: q,k,v [B,H,N,D] f16 + o [B,N,H*D] f16 = 4 x 16 MB = 64 MB
    const size_t ELEMS = (size_t)Bsz * H * Nseq * Dh;  // 8388608
    _Float16* qh = (_Float16*)d_ws;
    _Float16* kh = qh + ELEMS;
    _Float16* vh = kh + ELEMS;
    _Float16* oy = vh + ELEMS;

    // 1) QKV projection: (16384/32)*(1536/64) = 12288 wave-tiles, 8 waves/block
    qkv_gemm_kernel<<<dim3(12288 / 8), dim3(256), 0, stream>>>(
        x, qkv_w, qkv_b, qh, kh, vh);

    // 2) Fused attention: B*H*(N/64) = 2048 blocks, 306176 B dynamic LDS
    attn_fused_kernel<<<dim3(Bsz * H * (Nseq / 64)), dim3(256), SMEM_BYTES, stream>>>(
        qh, kh, vh, pe_w, pe_b, oy);

    // 3) Output projection: (16384/32)*(512/64) = 4096 wave-tiles
    proj_gemm_kernel<<<dim3(4096 / 8), dim3(256), 0, stream>>>(
        oy, proj_w, proj_b, out);
}
